// EPMoE_20358144983697
// MI455X (gfx1250) — matrix operand order repaired and checked by reading, not compiled, not run on hardware
//
#include <hip/hip_runtime.h>
#include <hip/hip_bf16.h>
#include <math.h>

// Problem dims (fixed by reference)
#define T_TOK 1024
#define H_DIM 2048
#define E_NUM 8
#define I_DIM 1408
#define N1_DIM (2 * I_DIM)   // 2816
#define R_ROWS (2 * T_TOK)   // total (token,slot) assignments = 2048

// Tiling
#define BM 64
#define BN 128
#define BK 32
#define LDP (BK + 8)         // padded LDS row (80B = 5x16B -> rows stay 16B aligned)

typedef unsigned short u16;
typedef __attribute__((ext_vector_type(16))) __bf16 v16bf;
typedef __attribute__((ext_vector_type(8)))  float  v8f;

union FragAB { uint4 q[2]; v16bf v; };

__device__ __forceinline__ u16 f2bf(float f) {
  unsigned u = __float_as_uint(f);
  u += 0x7FFFu + ((u >> 16) & 1u);   // round-to-nearest-even to bf16
  return (u16)(u >> 16);
}
__device__ __forceinline__ float bf2f(u16 h) {
  return __uint_as_float(((unsigned)h) << 16);
}
__device__ __forceinline__ void splitbf(float f, u16* hp, u16* lp) {
  u16 h = f2bf(f);
  *hp = h;
  *lp = f2bf(f - bf2f(h));           // residual captures next 8 mantissa bits
}

__device__ __forceinline__ v8f wmma_bf16(v16bf a, v16bf b, v8f c) {
  // D = A(16x32 bf16) * B(32x16 bf16) + C(16x16 f32)
  return __builtin_amdgcn_wmma_f32_16x16x32_bf16(false, a, false, b, (short)0, c,
                                                 false, false);
}

// ---------------- Router: softmax -> top2 -> renormalize -------------------
__global__ void moe_router(const float* __restrict__ logits,
                           int* __restrict__ topk_id, float* __restrict__ topk_w) {
  int t = blockIdx.x * blockDim.x + threadIdx.x;
  if (t >= T_TOK) return;
  float l[E_NUM];
#pragma unroll
  for (int i = 0; i < E_NUM; ++i) l[i] = logits[t * E_NUM + i];
  int i1 = 0; float m1 = l[0];
#pragma unroll
  for (int i = 1; i < E_NUM; ++i) if (l[i] > m1) { m1 = l[i]; i1 = i; }
  int i2 = -1; float m2 = -3.0e38f;
#pragma unroll
  for (int i = 0; i < E_NUM; ++i) if (i != i1 && l[i] > m2) { m2 = l[i]; i2 = i; }
  // renormalized top-2 softmax weights == exp ratio (stable)
  float e2 = __expf(m2 - m1);
  float inv = 1.0f / (1.0f + e2);
  topk_id[t * 2 + 0] = i1; topk_id[t * 2 + 1] = i2;
  topk_w[t * 2 + 0] = inv; topk_w[t * 2 + 1] = e2 * inv;
}

// --------- Deterministic bucketing: counts, offsets, compacted lists -------
__global__ void moe_fill(const int* __restrict__ topk_id, const float* __restrict__ topk_w,
                         int* __restrict__ token_map, int* __restrict__ pos_of,
                         float* __restrict__ row_w, int* __restrict__ counts,
                         int* __restrict__ offsets) {
  __shared__ int s_cnt[E_NUM];
  __shared__ int s_off[E_NUM];
  const int e = threadIdx.x;
  if (e < E_NUM) {
    int c = 0;
    for (int i = 0; i < R_ROWS; ++i) c += (topk_id[i] == e) ? 1 : 0;
    s_cnt[e] = c;
  }
  __syncthreads();
  if (threadIdx.x == 0) {
    int acc = 0;
    for (int i = 0; i < E_NUM; ++i) { s_off[i] = acc; acc += s_cnt[i]; }
  }
  __syncthreads();
  if (e < E_NUM) {
    counts[e] = s_cnt[e];
    offsets[e] = s_off[e];
    int pos = s_off[e];
    for (int i = 0; i < R_ROWS; ++i) {   // stable order over (token, slot)
      if (topk_id[i] == e) {
        token_map[pos] = i >> 1;
        pos_of[i] = pos;
        row_w[pos] = topk_w[i];
        pos++;
      }
    }
  }
}

// ---------- GEMM1: act[pos, 0:I] = silu(x W13_gate) * (x W13_up) -----------
__global__ __launch_bounds__(256) void moe_gemm1(
    const float* __restrict__ hidden, const float* __restrict__ w13,
    const int* __restrict__ token_map, const int* __restrict__ counts,
    const int* __restrict__ offsets, float* __restrict__ act) {
  const int e = blockIdx.z;
  const int cnt = counts[e];
  const int m0 = blockIdx.y * BM;
  if (m0 >= cnt) return;
  const int off = offsets[e];
  const int n0 = blockIdx.x * BN;

  __shared__ __align__(16) u16 Ah[BM][LDP],  Al[BM][LDP];
  __shared__ __align__(16) u16 Bgh[BN][LDP], Bgl[BN][LDP];
  __shared__ __align__(16) u16 Buh[BN][LDP], Bul[BN][LDP];
  __shared__ int s_tok[BM];

  const int tid = threadIdx.x;
  if (tid < BM) {
    int idx = off + m0 + tid;
    s_tok[tid] = token_map[idx < R_ROWS ? idx : (R_ROWS - 1)];
  }
  __syncthreads();

  const int wave  = tid >> 5;
  const int lane  = tid & 31;
  const int r0    = (wave & 3) * 16;   // 16-row strip within 64
  const int c0    = (wave >> 2) * 64;  // 64-col half within 128
  const int lhalf = lane >> 4;
  const int lidx  = lane & 15;

  v8f accg[4], accu[4];
#pragma unroll
  for (int t = 0; t < 4; ++t)
#pragma unroll
    for (int v = 0; v < 8; ++v) { accg[t][v] = 0.0f; accu[t][v] = 0.0f; }

  const float* Bbase = w13 + (size_t)e * H_DIM * N1_DIM + n0;

  for (int k0 = 0; k0 < H_DIM; k0 += BK) {
    // A tile: 64x32 fp32, gathered rows -> bf16 hi/lo in LDS
#pragma unroll
    for (int i = 0; i < 2; ++i) {
      int lin = tid + i * 256;           // 0..511 float4s
      int row = lin >> 3;
      int kq  = (lin & 7) << 2;
      float4 f = *(const float4*)(hidden + (size_t)s_tok[row] * H_DIM + k0 + kq);
      splitbf(f.x, &Ah[row][kq + 0], &Al[row][kq + 0]);
      splitbf(f.y, &Ah[row][kq + 1], &Al[row][kq + 1]);
      splitbf(f.z, &Ah[row][kq + 2], &Al[row][kq + 2]);
      splitbf(f.w, &Ah[row][kq + 3], &Al[row][kq + 3]);
    }
    // B tiles (gate + up): 32x128 fp32 each, stored transposed [n][k]
#pragma unroll
    for (int i = 0; i < 4; ++i) {
      int lin = tid + i * 256;           // 0..1023 float4s
      int kr  = lin >> 5;                // 0..31
      int cq  = (lin & 31) << 2;         // col*4
      const float* gp = Bbase + (size_t)(k0 + kr) * N1_DIM + cq;
      float4 g = *(const float4*)gp;
      float4 u = *(const float4*)(gp + I_DIM);
      if (k0 + BK < H_DIM) __builtin_prefetch(gp + (size_t)BK * N1_DIM);
      splitbf(g.x, &Bgh[cq + 0][kr], &Bgl[cq + 0][kr]);
      splitbf(g.y, &Bgh[cq + 1][kr], &Bgl[cq + 1][kr]);
      splitbf(g.z, &Bgh[cq + 2][kr], &Bgl[cq + 2][kr]);
      splitbf(g.w, &Bgh[cq + 3][kr], &Bgl[cq + 3][kr]);
      splitbf(u.x, &Buh[cq + 0][kr], &Bul[cq + 0][kr]);
      splitbf(u.y, &Buh[cq + 1][kr], &Bul[cq + 1][kr]);
      splitbf(u.z, &Buh[cq + 2][kr], &Bul[cq + 2][kr]);
      splitbf(u.w, &Buh[cq + 3][kr], &Bul[cq + 3][kr]);
    }
    __syncthreads();

    // A fragment: lane<16 -> K 0..15 ; lane>=16 -> K 16..31
    FragAB ah, al;
    {
      const int ar = r0 + lidx;
      const int kb = lhalf * 16;
      ah.q[0] = *(const uint4*)&Ah[ar][kb];
      ah.q[1] = *(const uint4*)&Ah[ar][kb + 8];
      al.q[0] = *(const uint4*)&Al[ar][kb];
      al.q[1] = *(const uint4*)&Al[ar][kb + 8];
    }
    const int kbb = lhalf * 16;          // B: lane<16 -> K0..15, lane>=16 -> K16..31
#pragma unroll
    for (int t = 0; t < 4; ++t) {
      const int bc = c0 + t * 16 + lidx;
      FragAB bh, bl;
      bh.q[0] = *(const uint4*)&Bgh[bc][kbb];
      bh.q[1] = *(const uint4*)&Bgh[bc][kbb + 8];
      bl.q[0] = *(const uint4*)&Bgl[bc][kbb];
      bl.q[1] = *(const uint4*)&Bgl[bc][kbb + 8];
      accg[t] = wmma_bf16(ah.v, bh.v, accg[t]);
      accg[t] = wmma_bf16(ah.v, bl.v, accg[t]);
      accg[t] = wmma_bf16(al.v, bh.v, accg[t]);
      bh.q[0] = *(const uint4*)&Buh[bc][kbb];
      bh.q[1] = *(const uint4*)&Buh[bc][kbb + 8];
      bl.q[0] = *(const uint4*)&Bul[bc][kbb];
      bl.q[1] = *(const uint4*)&Bul[bc][kbb + 8];
      accu[t] = wmma_bf16(ah.v, bh.v, accu[t]);
      accu[t] = wmma_bf16(ah.v, bl.v, accu[t]);
      accu[t] = wmma_bf16(al.v, bh.v, accu[t]);
    }
    __syncthreads();
  }

  // Epilogue: fused SiLU(gate) * up, written to compacted activation rows
#pragma unroll
  for (int t = 0; t < 4; ++t)
#pragma unroll
    for (int v = 0; v < 8; ++v) {
      int lrow = r0 + v + (lhalf << 3);   // C layout: M = vgpr + 8*(lane>=16)
      int m = m0 + lrow;
      if (m < cnt) {
        float g = accg[t][v];
        float u = accu[t][v];
        float s = g / (1.0f + __expf(-g));
        act[(size_t)(off + m) * I_DIM + (n0 + c0 + t * 16 + lidx)] = s * u;
      }
    }
}

// ---------- GEMM2: part[pos, 0:H] = route_w * (act W2[e]) ------------------
__global__ __launch_bounds__(256) void moe_gemm2(
    const float* __restrict__ act, const float* __restrict__ w2,
    const float* __restrict__ row_w, const int* __restrict__ counts,
    const int* __restrict__ offsets, float* __restrict__ part) {
  const int e = blockIdx.z;
  const int cnt = counts[e];
  const int m0 = blockIdx.y * BM;
  if (m0 >= cnt) return;
  const int off = offsets[e];
  const int n0 = blockIdx.x * BN;

  __shared__ __align__(16) u16 Ah[BM][LDP], Al[BM][LDP];
  __shared__ __align__(16) u16 Bh[BN][LDP], Bl[BN][LDP];
  __shared__ float s_w[BM];

  const int tid = threadIdx.x;
  if (tid < BM) {
    int idx = off + m0 + tid;
    s_w[tid] = row_w[idx < R_ROWS ? idx : (R_ROWS - 1)];
  }
  __syncthreads();

  const int wave  = tid >> 5;
  const int lane  = tid & 31;
  const int r0    = (wave & 3) * 16;
  const int c0    = (wave >> 2) * 64;
  const int lhalf = lane >> 4;
  const int lidx  = lane & 15;

  v8f acc[4];
#pragma unroll
  for (int t = 0; t < 4; ++t)
#pragma unroll
    for (int v = 0; v < 8; ++v) acc[t][v] = 0.0f;

  const float* Bbase = w2 + (size_t)e * I_DIM * H_DIM + n0;

  for (int k0 = 0; k0 < I_DIM; k0 += BK) {
#pragma unroll
    for (int i = 0; i < 2; ++i) {
      int lin = tid + i * 256;
      int row = lin >> 3;
      int kq  = (lin & 7) << 2;
      int gr  = off + m0 + row;
      gr = gr < R_ROWS ? gr : (R_ROWS - 1);
      float4 f = *(const float4*)(act + (size_t)gr * I_DIM + k0 + kq);
      splitbf(f.x, &Ah[row][kq + 0], &Al[row][kq + 0]);
      splitbf(f.y, &Ah[row][kq + 1], &Al[row][kq + 1]);
      splitbf(f.z, &Ah[row][kq + 2], &Al[row][kq + 2]);
      splitbf(f.w, &Ah[row][kq + 3], &Al[row][kq + 3]);
    }
#pragma unroll
    for (int i = 0; i < 4; ++i) {
      int lin = tid + i * 256;
      int kr  = lin >> 5;
      int cq  = (lin & 31) << 2;
      const float* gp = Bbase + (size_t)(k0 + kr) * H_DIM + cq;
      float4 b = *(const float4*)gp;
      if (k0 + BK < I_DIM) __builtin_prefetch(gp + (size_t)BK * H_DIM);
      splitbf(b.x, &Bh[cq + 0][kr], &Bl[cq + 0][kr]);
      splitbf(b.y, &Bh[cq + 1][kr], &Bl[cq + 1][kr]);
      splitbf(b.z, &Bh[cq + 2][kr], &Bl[cq + 2][kr]);
      splitbf(b.w, &Bh[cq + 3][kr], &Bl[cq + 3][kr]);
    }
    __syncthreads();

    FragAB ah, al;
    {
      const int ar = r0 + lidx;
      const int kb = lhalf * 16;
      ah.q[0] = *(const uint4*)&Ah[ar][kb];
      ah.q[1] = *(const uint4*)&Ah[ar][kb + 8];
      al.q[0] = *(const uint4*)&Al[ar][kb];
      al.q[1] = *(const uint4*)&Al[ar][kb + 8];
    }
    const int kbb = lhalf * 16;
#pragma unroll
    for (int t = 0; t < 4; ++t) {
      const int bc = c0 + t * 16 + lidx;
      FragAB bh, bl;
      bh.q[0] = *(const uint4*)&Bh[bc][kbb];
      bh.q[1] = *(const uint4*)&Bh[bc][kbb + 8];
      bl.q[0] = *(const uint4*)&Bl[bc][kbb];
      bl.q[1] = *(const uint4*)&Bl[bc][kbb + 8];
      acc[t] = wmma_bf16(ah.v, bh.v, acc[t]);
      acc[t] = wmma_bf16(ah.v, bl.v, acc[t]);
      acc[t] = wmma_bf16(al.v, bh.v, acc[t]);
    }
    __syncthreads();
  }

#pragma unroll
  for (int t = 0; t < 4; ++t)
#pragma unroll
    for (int v = 0; v < 8; ++v) {
      int lrow = r0 + v + (lhalf << 3);
      int m = m0 + lrow;
      if (m < cnt) {
        part[(size_t)(off + m) * H_DIM + (n0 + c0 + t * 16 + lidx)] =
            s_w[lrow] * acc[t][v];
      }
    }
}

// ---------- Combine the two expert-slot contributions per token ------------
__global__ void moe_combine(const float* __restrict__ part,
                            const int* __restrict__ pos_of,
                            float* __restrict__ out) {
  const int t = blockIdx.x;
  const int p0 = pos_of[t * 2 + 0];
  const int p1 = pos_of[t * 2 + 1];
  const float4* a = (const float4*)(part + (size_t)p0 * H_DIM);
  const float4* b = (const float4*)(part + (size_t)p1 * H_DIM);
  float4* o = (float4*)(out + (size_t)t * H_DIM);
  for (int i = threadIdx.x; i < H_DIM / 4; i += blockDim.x) {
    float4 x = a[i], y = b[i];
    o[i] = make_float4(x.x + y.x, x.y + y.y, x.z + y.z, x.w + y.w);
  }
}

extern "C" void kernel_launch(void* const* d_in, const int* in_sizes, int n_in,
                              void* d_out, int out_size, void* d_ws, size_t ws_size,
                              hipStream_t stream) {
  (void)in_sizes; (void)n_in; (void)out_size; (void)ws_size;
  const float* hidden = (const float*)d_in[0];
  const float* logits = (const float*)d_in[1];
  const float* w13    = (const float*)d_in[2];
  const float* w2     = (const float*)d_in[3];
  float* out = (float*)d_out;

  char* ws = (char*)d_ws;
  int*   topk_id   = (int*)(ws + 0);          // 2T ints
  float* topk_w    = (float*)(ws + 8192);     // 2T floats
  int*   token_map = (int*)(ws + 16384);      // 2T ints
  int*   pos_of    = (int*)(ws + 24576);      // 2T ints
  float* row_w     = (float*)(ws + 32768);    // 2T floats
  int*   counts    = (int*)(ws + 40960);      // E ints
  int*   offsets   = (int*)(ws + 41024);      // E ints
  float* act       = (float*)(ws + 65536);                                  // R x I
  float* part      = (float*)(ws + 65536 + (size_t)R_ROWS * I_DIM * 4);     // R x H

  moe_router <<<dim3(T_TOK / 256), dim3(256), 0, stream>>>(logits, topk_id, topk_w);
  moe_fill   <<<dim3(1), dim3(32), 0, stream>>>(topk_id, topk_w, token_map, pos_of,
                                                row_w, counts, offsets);
  moe_gemm1  <<<dim3(I_DIM / BN, T_TOK / BM, E_NUM), dim3(256), 0, stream>>>(
      hidden, w13, token_map, counts, offsets, act);
  moe_gemm2  <<<dim3(H_DIM / BN, T_TOK / BM, E_NUM), dim3(256), 0, stream>>>(
      act, w2, row_w, counts, offsets, part);
  moe_combine<<<dim3(T_TOK), dim3(256), 0, stream>>>(part, pos_of, out);
}
